// GAT_Classifier_29755533426834
// MI455X (gfx1250) — matrix operand order repaired and checked by reading, not compiled, not run on hardware
//
#include <hip/hip_runtime.h>

// GAT forward for MI455X (gfx1250, wave32, WMMA f32<-f16 16x16x32).
// N=4096 nodes, F=512, H=8 heads, D=64, layer2: D2=16 classes.
// Softmax is computed in the exp2 domain: score vectors are pre-scaled by
// log2(e) (valid since LeakyReLU commutes with positive scaling), so the hot
// aggregation loop needs exactly one v_exp_f32 per attention entry.

#define NN    4096
#define NFEAT 512
#define NHID  512
#define NH    8
#define DH    64
#define NCLS  16

#define LOG2E      1.4426950408889634f
#define NEG_FILL2  (-1.2984016e16f)   // -9e15 * log2(e)

typedef _Float16 h16 __attribute__((ext_vector_type(16)));
typedef _Float16 h8  __attribute__((ext_vector_type(8)));
typedef _Float16 h2  __attribute__((ext_vector_type(2)));
typedef __fp16   fp16x2 __attribute__((ext_vector_type(2)));
typedef float    f8v __attribute__((ext_vector_type(8)));
typedef float    f4v __attribute__((ext_vector_type(4)));
typedef int      i4v __attribute__((ext_vector_type(4)));

union Frag16 { h16 v; h8 h[2]; h2 p[8]; };
union Out8   { h8  v; h2 p[4]; };

__device__ __forceinline__ float fast_exp2(float x) {
#if __has_builtin(__builtin_amdgcn_exp2f)
    return __builtin_amdgcn_exp2f(x);          // v_exp_f32
#else
    return __expf(x * 0.6931471805599453f);    // exp(ln2*x) == 2^x
#endif
}

__device__ __forceinline__ h2 pk2(float a, float b) {
#if __has_builtin(__builtin_amdgcn_cvt_pkrtz)
    fp16x2 t = __builtin_amdgcn_cvt_pkrtz(a, b);   // v_cvt_pk_rtz_f16_f32
    return __builtin_bit_cast(h2, t);
#else
    h2 r; r[0] = (_Float16)a; r[1] = (_Float16)b; return r;
#endif
}

__device__ __forceinline__ float elu1(float x) {
    return x > 0.f ? x : (__expf(x) - 1.f);
}

// ---------------- conversion / packing kernels ----------------

__global__ void k_cvt_f16(const float* __restrict__ in, _Float16* __restrict__ out, int n) {
    int i = blockIdx.x * blockDim.x + threadIdx.x;
    if (i < n) out[i] = (_Float16)in[i];
}

// W[h][f][d] (f32) -> Wc[c][f] (f16, col-major over output column c = h*64+d)
__global__ void k_pack_W(const float* __restrict__ W, _Float16* __restrict__ Wc, int n) {
    int i = blockIdx.x * blockDim.x + threadIdx.x;
    if (i < n) {
        int c = i >> 9, f = i & 511;
        Wc[i] = (_Float16)W[((c >> 6) << 15) + (f << 6) + (c & 63)];
    }
}

// Wo[f][c] (f32, 512x16) -> WoC[c][f] (f16, 16x512)
__global__ void k_pack_Wo(const float* __restrict__ Wo, _Float16* __restrict__ WoC, int n) {
    int i = blockIdx.x * blockDim.x + threadIdx.x;
    if (i < n) {
        int c = i >> 9, f = i & 511;
        WoC[i] = (_Float16)Wo[f * 16 + c];
    }
}

// ---------------- WMMA GEMM: CT[c][m] = sum_k A[m][k]*Bc[c][k] ----------------
// A row-major f16 [M][K]; Bc col-major f16 [Ncols][K]; CT f16 [Ncols][M].
// One wave computes one 16x16 tile. K multiple of 32.

__global__ void k_gemm_wmma(const _Float16* __restrict__ A,
                            const _Float16* __restrict__ Bc,
                            _Float16* __restrict__ CT,
                            int M, int N, int K) {
    int lane = threadIdx.x & 31;
    int gw   = blockIdx.x * (blockDim.x >> 5) + (threadIdx.x >> 5);
    int mTiles = M >> 4, nTiles = N >> 4;
    if (gw >= mTiles * nTiles) return;
    int mt = gw % mTiles, nt = gw / mTiles;

    int half = lane >> 4, ln = lane & 15;
    int row = mt * 16 + ln;          // A row this lane feeds
    int col = nt * 16 + ln;          // B column this lane feeds
    int kbA = half * 8;              // A: halves hold K {0..7,16..23} / {8..15,24..31}
    int kbB = half * 8;              // B: halves hold K {0..7,16..23} / {8..15,24..31}

    f8v acc = {};
    const _Float16* ar = A  + (size_t)row * K;
    const _Float16* br = Bc + (size_t)col * K;
    for (int kc = 0; kc < K; kc += 32) {
        Frag16 a, b;
        a.h[0] = *(const h8*)(ar + kc + kbA);
        a.h[1] = *(const h8*)(ar + kc + kbA + 16);
        b.h[0] = *(const h8*)(br + kc + kbB);
        b.h[1] = *(const h8*)(br + kc + kbB + 16);
        acc = __builtin_amdgcn_wmma_f32_16x16x32_f16(false, a.v, false, b.v,
                                                     (short)0, acc, false, false);
    }
    // C layout: lane half picks rows i+8*half, column = ln. Rows contiguous -> h8 store.
    Out8 o;
#pragma unroll
    for (int q = 0; q < 4; ++q) o.p[q] = pk2(acc[2 * q], acc[2 * q + 1]);
    *(h8*)(CT + (size_t)col * M + mt * 16 + half * 8) = o.v;
}

// ---------------- attention score vectors: fs[h][n], fd[h][n] ----------------
// CT = [H*D][Nn] f16 (transposed Wh). a vectors f32 [H*D].
// Outputs are pre-scaled by log2(e) so softmax runs in the exp2 domain.

__global__ void k_fvec(const _Float16* __restrict__ CT,
                       const float* __restrict__ a_src,
                       const float* __restrict__ a_dst,
                       float* __restrict__ fs, float* __restrict__ fd,
                       int H, int D, int Nn) {
    int idx = blockIdx.x * blockDim.x + threadIdx.x;
    int n = idx % Nn, h = idx / Nn;
    if (h >= H) return;
    float s = 0.f, t = 0.f;
    for (int d = 0; d < D; ++d) {
        float w = (float)CT[(size_t)(h * D + d) * Nn + n];
        s += w * a_src[h * D + d];
        t += w * a_dst[h * D + d];
    }
    fs[h * Nn + n] = s * LOG2E;
    fd[h * Nn + n] = t * LOG2E;
}

// ---------------- softmax stats: row max m (exp2 domain) and 1/l -------------
// One wave per (head,row). Online (m,l) per lane, then shfl_xor tree merge.

__global__ void k_stats(const int* __restrict__ adj,
                        const float* __restrict__ fs, const float* __restrict__ fd,
                        float* __restrict__ mm, float* __restrict__ il,
                        int H, int Nn) {
    int lane = threadIdx.x & 31;
    int gw = blockIdx.x * (blockDim.x >> 5) + (threadIdx.x >> 5);
    int row = gw % Nn, h = gw / Nn;
    if (h >= H) return;
    float fsm = fs[h * Nn + row];
    const float* fdh = fd + (size_t)h * Nn;
    const int* arow = adj + (size_t)row * Nn;
    float m = -3.0e38f, l = 0.f;
    for (int j = lane; j < Nn; j += 32) {
        float v = fsm + fdh[j];
        v = v > 0.f ? v : 0.2f * v;            // LeakyReLU(0.2) (scale-invariant)
        v = arow[j] > 0 ? v : NEG_FILL2;       // pyGAT mask fill (scaled)
        float mn = fmaxf(m, v);
        l = l * fast_exp2(m - mn) + fast_exp2(v - mn);
        m = mn;
    }
#pragma unroll
    for (int off = 16; off >= 1; off >>= 1) {
        float m2 = __shfl_xor(m, off, 32);
        float l2 = __shfl_xor(l, off, 32);
        float mn = fmaxf(m, m2);
        l = l * fast_exp2(m - mn) + l2 * fast_exp2(m2 - mn);
        m = mn;
    }
    if (lane == 0) {
        mm[h * Nn + row] = m;
        il[h * Nn + row] = l > 0.f ? 1.0f / l : 0.f;
    }
}

// ---------------- fused attention aggregation (WMMA streaming GEMM) ----------
// out_row_tile[16 x (NT*16)] = sum_j p[16x32] * V, p recomputed in registers:
// p = exp2(leaky2(fs'+fd') - m') * (1/l), one v_exp_f32 per entry.
// NT=4,FINAL=0: layer1 -> elu -> h1 f16 [Nn][512] at column h*64.
// NT=1,FINAL=1: layer2 -> elu(elu) -> f32 out [Nn][16].

template <int NT, bool FINAL>
__global__ void k_agg(const int* __restrict__ adj,
                      const float* __restrict__ fs, const float* __restrict__ fd,
                      const float* __restrict__ mm, const float* __restrict__ il,
                      const _Float16* __restrict__ VT,   // [H*NT*16][Nn]
                      int H, int Nn,
                      _Float16* __restrict__ h1out, float* __restrict__ fout) {
    int lane = threadIdx.x & 31;
    int gw = blockIdx.x * (blockDim.x >> 5) + (threadIdx.x >> 5);
    int mTiles = Nn >> 4;
    int h = gw % H, mt = gw / H;
    if (mt >= mTiles) return;

    int half = lane >> 4, ln = lane & 15;
    int row = mt * 16 + ln;
    float fsm = fs[h * Nn + row];
    float mmv = mm[h * Nn + row];
    float ilv = il[h * Nn + row];
    int b8 = half * 8;                       // A-fragment K offset for this half-wave
    int kb = half * 8;                       // B-fragment K offset (same as A)

    const int*   arow = adj + (size_t)row * Nn + b8;
    const float* fdh  = fd + (size_t)h * Nn + b8;
    const _Float16* Vb = VT + (size_t)h * (NT * 16) * Nn;

    f8v acc[NT] = {};
    for (int jc = 0; jc < Nn; jc += 32) {
        i4v a0 = *(const i4v*)(arow + jc);
        i4v a1 = *(const i4v*)(arow + jc + 4);
        i4v a2 = *(const i4v*)(arow + jc + 16);
        i4v a3 = *(const i4v*)(arow + jc + 20);
        __builtin_prefetch((const void*)(arow + jc + 32), 0, 1);
        f4v d0 = *(const f4v*)(fdh + jc);
        f4v d1 = *(const f4v*)(fdh + jc + 4);
        f4v d2 = *(const f4v*)(fdh + jc + 16);
        f4v d3 = *(const f4v*)(fdh + jc + 20);

        float pv[16];
#pragma unroll
        for (int e = 0; e < 8; ++e) {
            float v = fsm + ((e < 4) ? d0[e] : d1[e - 4]);
            v = v > 0.f ? v : 0.2f * v;
            int ab = (e < 4) ? a0[e] : a1[e - 4];
            v = ab > 0 ? v : NEG_FILL2;
            pv[e] = fast_exp2(v - mmv) * ilv;

            float w = fsm + ((e < 4) ? d2[e] : d3[e - 4]);
            w = w > 0.f ? w : 0.2f * w;
            int ab2 = (e < 4) ? a2[e] : a3[e - 4];
            w = ab2 > 0 ? w : NEG_FILL2;
            pv[8 + e] = fast_exp2(w - mmv) * ilv;
        }
        Frag16 p;
#pragma unroll
        for (int q = 0; q < 8; ++q) p.p[q] = pk2(pv[2 * q], pv[2 * q + 1]);

#pragma unroll
        for (int s = 0; s < NT; ++s) {
            Frag16 b;
            const _Float16* vp = Vb + (size_t)(s * 16 + ln) * Nn + jc + kb;
            b.h[0] = *(const h8*)vp;
            b.h[1] = *(const h8*)(vp + 16);
            acc[s] = __builtin_amdgcn_wmma_f32_16x16x32_f16(false, p.v, false, b.v,
                                                            (short)0, acc[s], false, false);
        }
    }

#pragma unroll
    for (int s = 0; s < NT; ++s) {
#pragma unroll
        for (int i = 0; i < 8; ++i) {
            int ro = mt * 16 + i + half * 8;
            float v = elu1(acc[s][i]);
            if (FINAL) {
                fout[(size_t)ro * 16 + ln] = elu1(v);       // outer F.elu
            } else {
                h1out[(size_t)ro * NHID + h * DH + s * 16 + ln] = (_Float16)v;
            }
        }
    }
}

// ---------------- host side ----------------

extern "C" void kernel_launch(void* const* d_in, const int* in_sizes, int n_in,
                              void* d_out, int out_size, void* d_ws, size_t ws_size,
                              hipStream_t stream) {
    const float* x      = (const float*)d_in[0];
    const int*   adj    = (const int*)d_in[1];
    const float* W      = (const float*)d_in[2];
    const float* a_src  = (const float*)d_in[3];
    const float* a_dst  = (const float*)d_in[4];
    const float* Wo     = (const float*)d_in[5];
    const float* ao_src = (const float*)d_in[6];
    const float* ao_dst = (const float*)d_in[7];
    float* out = (float*)d_out;

    char* ws = (char*)d_ws;
    size_t off = 0;
    auto alloc = [&](size_t bytes) { void* p = ws + off; off = (off + bytes + 255) & ~(size_t)255; return p; };
    _Float16* xh   = (_Float16*)alloc((size_t)NN * NFEAT * 2);       // x in f16
    _Float16* Wc   = (_Float16*)alloc((size_t)NHID * NFEAT * 2);     // W col-major f16
    _Float16* WoC  = (_Float16*)alloc((size_t)NCLS * NHID * 2);      // Wo col-major f16
    _Float16* WhT  = (_Float16*)alloc((size_t)NHID * NN * 2);        // Wh transposed [c][n]
    float*    fs1  = (float*)alloc((size_t)NH * NN * 4);
    float*    fd1  = (float*)alloc((size_t)NH * NN * 4);
    float*    mm1  = (float*)alloc((size_t)NH * NN * 4);
    float*    il1  = (float*)alloc((size_t)NH * NN * 4);
    _Float16* h1   = (_Float16*)alloc((size_t)NN * NHID * 2);        // layer1 out f16
    _Float16* W2T  = (_Float16*)alloc((size_t)NCLS * NN * 2);        // Wh2 transposed
    float*    fs2  = (float*)alloc((size_t)NN * 4);
    float*    fd2  = (float*)alloc((size_t)NN * 4);
    float*    mm2  = (float*)alloc((size_t)NN * 4);
    float*    il2  = (float*)alloc((size_t)NN * 4);

    const int T = 256;  // 8 waves / block

    // 1) precision conversion + weight packing
    k_cvt_f16<<<(NN * NFEAT + T - 1) / T, T, 0, stream>>>(x, xh, NN * NFEAT);
    k_pack_W <<<(NHID * NFEAT + T - 1) / T, T, 0, stream>>>(W, Wc, NHID * NFEAT);
    k_pack_Wo<<<(NCLS * NFEAT + T - 1) / T, T, 0, stream>>>(Wo, WoC, NCLS * NHID);

    // 2) layer-1 feature GEMM: WhT[c][n] (256*32 tiles = 8192 waves)
    k_gemm_wmma<<<(256 * 32) / 8, T, 0, stream>>>(xh, Wc, WhT, NN, NHID, NFEAT);

    // 3) score vectors (log2e-scaled), softmax stats (wave per (h,row))
    k_fvec <<<(NH * NN + T - 1) / T, T, 0, stream>>>(WhT, a_src, a_dst, fs1, fd1, NH, DH, NN);
    k_stats<<<(NH * NN) / 8, T, 0, stream>>>(adj, fs1, fd1, mm1, il1, NH, NN);

    // 4) fused masked-softmax attention aggregation + ELU (wave per (head, row-tile))
    k_agg<4, false><<<(NH * 256) / 8, T, 0, stream>>>(adj, fs1, fd1, mm1, il1, WhT,
                                                      NH, NN, h1, nullptr);

    // 5) layer-2 GEMM: W2T[c][n] (256 tiles)
    k_gemm_wmma<<<256 / 8, T, 0, stream>>>(h1, WoC, W2T, NN, NCLS, NHID);

    // 6) layer-2 scores, stats, final aggregation with elu(elu(.))
    k_fvec <<<(NN + T - 1) / T, T, 0, stream>>>(W2T, ao_src, ao_dst, fs2, fd2, 1, NCLS, NN);
    k_stats<<<NN / 8, T, 0, stream>>>(adj, fs2, fd2, mm2, il2, 1, NN);
    k_agg<1, true><<<256 / 8, T, 0, stream>>>(adj, fs2, fd2, mm2, il2, W2T,
                                              1, NN, nullptr, out);

    (void)in_sizes; (void)n_in; (void)out_size; (void)ws_size;
}
